// MultiHeadedAttention_52037823758722
// MI455X (gfx1250) — matrix operand order repaired and checked by reading, not compiled, not run on hardware
//
#include <hip/hip_runtime.h>
#include <math.h>

typedef __attribute__((ext_vector_type(16))) _Float16 v16h;
typedef __attribute__((ext_vector_type(8)))  _Float16 v8h;
typedef __attribute__((ext_vector_type(4)))  _Float16 v4h;
typedef __attribute__((ext_vector_type(8)))  float    v8f;
typedef __attribute__((ext_vector_type(4)))  float    v4f;

#define B_  32
#define L_  1024
#define D_  512
#define H_  8
#define DK_ 64
#define W_  5
#define M_  (B_ * L_)   // 32768 rows
#define WPACK_ELEMS (D_ * D_)   // 262144 halves per packed weight matrix

// WMMA fragment layouts (CDNA5 ISA 7.12.2, wave32):
//  A (16x32 f16): elem e of v16h -> M = lane&15, K = 16*(e>>3) + 8*(lane>>4) + (e&7)
//  B (32x16 f16): elem e         -> N = lane&15, K = 16*(e>>3) + 8*(lane>>4) + (e&7)
//  C/D (16x16 f32): elem e of v8f-> N = lane&15, M = e + 8*(lane>>4)

// ---------------------------------------------------------------------------
// One-time weight pack: W[512,512] f32 (row major) -> f16 B-fragment tiles.
// Flat layout: (((kchunk * 32) + ntile) * 32 + lane) * 16 + e
// ---------------------------------------------------------------------------
__global__ void __launch_bounds__(256)
k_pack_w(const float* __restrict__ Wm, _Float16* __restrict__ Wp)
{
    const int gid  = blockIdx.x * 256 + threadIdx.x;   // 0 .. 16383
    const int c5   = gid >> 10;          // kchunk 0..15
    const int nt   = (gid >> 5) & 31;    // ntile  0..31
    const int lane = gid & 31;
    const int hf   = lane >> 4;
    const int l16  = lane & 15;

    v16h frag;
    #pragma unroll
    for (int e = 0; e < 16; ++e)
        frag[e] = (_Float16)Wm[(size_t)((c5 << 5) + ((e >> 3) << 4) + (hf << 3) + (e & 7)) * D_ + (nt << 4) + l16];
    *(v16h*)&Wp[(size_t)gid * 16] = frag;
}

// ---------------------------------------------------------------------------
// GEMM: Y[f16] = X[f32] @ W + b  (packed W). Workgroup = 32 rows; each wave
// computes a 32x64 tile (2 row-tiles x 4 col-tiles = 8 WMMA / K-step sharing
// 4 B fragments). Double-buffered LDS A tile, clamped unconditional prefetch,
// one barrier per K-step.
// ---------------------------------------------------------------------------
__global__ void __launch_bounds__(256)
k_linear_f32_f16(const float* __restrict__ X, const _Float16* __restrict__ Wp,
                 const float* __restrict__ bias, _Float16* __restrict__ Y)
{
    __shared__ _Float16 As[2][32][40];
    const int tid  = threadIdx.x;
    const int wave = tid >> 5;
    const int lane = tid & 31;
    const int hf   = lane >> 4;
    const int l16  = lane & 15;
    const int rowbase = blockIdx.x << 5;

    // staging map: thread -> (row sr, cols sc..sc+3)
    const int sr = tid >> 3;
    const int sc = (tid & 7) << 2;
    const float* xp = &X[(size_t)(rowbase + sr) * D_ + sc];

    const _Float16* wbase = Wp + (((size_t)wave << 7) + lane) * 16;

    v8f acc[2][4];
    #pragma unroll
    for (int rt = 0; rt < 2; ++rt)
        #pragma unroll
        for (int t = 0; t < 4; ++t)
            #pragma unroll
            for (int e = 0; e < 8; ++e) acc[rt][t][e] = 0.f;

    {   // stage K-step 0
        const v4f xv = *(const v4f*)xp;
        v4h hv;
        #pragma unroll
        for (int i = 0; i < 4; ++i) hv[i] = (_Float16)xv[i];
        *(v4h*)&As[0][sr][sc] = hv;
    }
    __syncthreads();

    for (int step = 0; step < 16; ++step) {
        const int p = step & 1;
        const int nxt = (step < 15) ? step + 1 : 15;          // clamped prefetch
        const v4f xv = *(const v4f*)(xp + (nxt << 5));

        v16h a[2];
        #pragma unroll
        for (int rt = 0; rt < 2; ++rt)
            #pragma unroll
            for (int e = 0; e < 16; ++e)
                a[rt][e] = As[p][(rt << 4) + l16][((e >> 3) << 4) + (hf << 3) + (e & 7)];

        const _Float16* wq = wbase + ((size_t)step << 14);
        #pragma unroll
        for (int t = 0; t < 4; ++t) {
            const v16h bf = *(const v16h*)(wq + (t << 9));
            #pragma unroll
            for (int rt = 0; rt < 2; ++rt)
                acc[rt][t] = __builtin_amdgcn_wmma_f32_16x16x32_f16(
                    false, a[rt], false, bf, (short)0, acc[rt][t], false, false);
        }

        v4h hv;
        #pragma unroll
        for (int i = 0; i < 4; ++i) hv[i] = (_Float16)xv[i];
        *(v4h*)&As[p ^ 1][sr][sc] = hv;                        // dead on last iter
        __syncthreads();
    }

    #pragma unroll
    for (int rt = 0; rt < 2; ++rt)
        #pragma unroll
        for (int t = 0; t < 4; ++t) {
            const int n = (wave << 6) + (t << 4) + l16;
            const float bv = bias[n];
            #pragma unroll
            for (int e = 0; e < 8; ++e) {
                const int row = rowbase + (rt << 4) + e + (hf << 3);
                Y[(size_t)row * D_ + n] = (_Float16)(acc[rt][t][e] + bv);
            }
        }
}

// ---------------------------------------------------------------------------
// GEMM: Y[f32] = X[f16] @ W + b  (output projection -> d_out), packed W.
// ---------------------------------------------------------------------------
__global__ void __launch_bounds__(256)
k_proj_f16_f32(const _Float16* __restrict__ X, const _Float16* __restrict__ Wp,
               const float* __restrict__ bias, float* __restrict__ Y)
{
    __shared__ _Float16 As[2][32][40];
    const int tid  = threadIdx.x;
    const int wave = tid >> 5;
    const int lane = tid & 31;
    const int hf   = lane >> 4;
    const int l16  = lane & 15;
    const int rowbase = blockIdx.x << 5;

    const int sr = tid >> 3;
    const int sc = (tid & 7) << 2;
    const _Float16* xp = &X[(size_t)(rowbase + sr) * D_ + sc];

    const _Float16* wbase = Wp + (((size_t)wave << 7) + lane) * 16;

    v8f acc[2][4];
    #pragma unroll
    for (int rt = 0; rt < 2; ++rt)
        #pragma unroll
        for (int t = 0; t < 4; ++t)
            #pragma unroll
            for (int e = 0; e < 8; ++e) acc[rt][t][e] = 0.f;

    {
        *(v4h*)&As[0][sr][sc] = *(const v4h*)xp;
    }
    __syncthreads();

    for (int step = 0; step < 16; ++step) {
        const int p = step & 1;
        const int nxt = (step < 15) ? step + 1 : 15;
        const v4h xv = *(const v4h*)(xp + (nxt << 5));

        v16h a[2];
        #pragma unroll
        for (int rt = 0; rt < 2; ++rt)
            #pragma unroll
            for (int e = 0; e < 16; ++e)
                a[rt][e] = As[p][(rt << 4) + l16][((e >> 3) << 4) + (hf << 3) + (e & 7)];

        const _Float16* wq = wbase + ((size_t)step << 14);
        #pragma unroll
        for (int t = 0; t < 4; ++t) {
            const v16h bf = *(const v16h*)(wq + (t << 9));
            #pragma unroll
            for (int rt = 0; rt < 2; ++rt)
                acc[rt][t] = __builtin_amdgcn_wmma_f32_16x16x32_f16(
                    false, a[rt], false, bf, (short)0, acc[rt][t], false, false);
        }

        *(v4h*)&As[p ^ 1][sr][sc] = xv;
        __syncthreads();
    }

    #pragma unroll
    for (int rt = 0; rt < 2; ++rt)
        #pragma unroll
        for (int t = 0; t < 4; ++t) {
            const int n = (wave << 6) + (t << 4) + l16;
            const float bv = bias[n];
            #pragma unroll
            for (int e = 0; e < 8; ++e) {
                const int row = rowbase + (rt << 4) + e + (hf << 3);
                Y[(size_t)row * D_ + n] = acc[rt][t][e] + bv;
            }
        }
}

// ---------------------------------------------------------------------------
// Sliding-window local mix: one wave per (b,l); lane owns 16 dims.
// ---------------------------------------------------------------------------
__global__ void __launch_bounds__(256)
k_local_mix(const _Float16* __restrict__ Xl, const float* __restrict__ b0,
            _Float16* __restrict__ Out)
{
    const int wave = threadIdx.x >> 5;
    const int lane = threadIdx.x & 31;
    const int row  = (blockIdx.x << 3) + wave;   // flat b*L + l
    const int l    = row & (L_ - 1);
    const int d0   = lane << 4;

    float win[W_][16];
    #pragma unroll
    for (int j = 0; j < W_; ++j) {
        const int pos = l - (W_ - 1) + j;
        if (pos < 0) {
            #pragma unroll
            for (int i = 0; i < 16; ++i) win[j][i] = b0[d0 + i];
        } else {
            const _Float16* p = &Xl[(size_t)(row - (W_ - 1) + j) * D_ + d0];
            #pragma unroll
            for (int i = 0; i < 16; ++i) win[j][i] = (float)p[i];
        }
    }
    float s[W_];
    #pragma unroll
    for (int j = 0; j < W_; ++j) {
        float p = 0.f;
        #pragma unroll
        for (int i = 0; i < 16; ++i) p += win[W_ - 1][i] * win[j][i];
        #pragma unroll
        for (int m = 16; m >= 1; m >>= 1) p += __shfl_xor(p, m, 32);
        s[j] = p * 0.044194173824159216f;  // 1/sqrt(512)
    }
    float mx = s[0];
    #pragma unroll
    for (int j = 1; j < W_; ++j) mx = fmaxf(mx, s[j]);
    float sum = 0.f;
    #pragma unroll
    for (int j = 0; j < W_; ++j) { s[j] = __expf(s[j] - mx); sum += s[j]; }
    const float inv = 1.f / sum;
    #pragma unroll
    for (int i = 0; i < 16; ++i) {
        float o = 0.f;
        #pragma unroll
        for (int j = 0; j < W_; ++j) o += s[j] * win[j][i];
        Out[(size_t)row * D_ + d0 + i] = (_Float16)(o * inv);
    }
}

// ---------------------------------------------------------------------------
// Flash attention, one workgroup = 8 waves = 128 query rows of one (b,h).
// Double-buffered K (row-major) and V (transposed) tiles in LDS; next tile
// prefetched into registers (clamped, unconditional); ONE barrier per tile.
// P relayout uses per-wave LDS; same-wave DS ops are in-order (no barrier).
// ---------------------------------------------------------------------------
__global__ void __launch_bounds__(256)
k_flash_attn(const _Float16* __restrict__ Q, const _Float16* __restrict__ K,
             const _Float16* __restrict__ V, _Float16* __restrict__ O)
{
    __shared__ _Float16 Ks[2][32][72];     // [buf][key][dv]
    __shared__ _Float16 Vt[2][64][40];     // [buf][dv][key]  (transposed)
    __shared__ _Float16 Ps[8][16][40];     // per-wave P staging

    const int tid  = threadIdx.x;
    const int wave = tid >> 5;
    const int lane = tid & 31;
    const int hf   = lane >> 4;
    const int l16  = lane & 15;

    const int qblk = blockIdx.x & 7;
    const int h    = (blockIdx.x >> 3) & 7;
    const int b    = blockIdx.x >> 6;

    const int qbase = (qblk << 7) + (wave << 4);
    const size_t seqbase = (size_t)b * L_;
    const int hd = h * DK_;

    // cooperative-load map: thread -> (key row r, dv cols c..c+7)
    const int r = tid >> 3;           // 0..31
    const int c = (tid & 7) << 3;     // 0..56
    const _Float16* kp = &K[(seqbase + r) * D_ + hd + c];
    const _Float16* vp = &V[(seqbase + r) * D_ + hd + c];

    v16h aq[2];
    #pragma unroll
    for (int kc = 0; kc < 2; ++kc)
        #pragma unroll
        for (int e = 0; e < 16; ++e) {
            const int d = (kc << 5) + (hf << 4) + e;
            aq[kc][e] = Q[(seqbase + qbase + l16) * D_ + hd + d];
        }

    v8f acc[4];
    float mrow[8], lrow[8];
    #pragma unroll
    for (int e = 0; e < 8; ++e) { mrow[e] = -1e30f; lrow[e] = 0.f; }
    #pragma unroll
    for (int t = 0; t < 4; ++t)
        #pragma unroll
        for (int e = 0; e < 8; ++e) acc[t][e] = 0.f;

    {   // stage tile 0
        *(v8h*)&Ks[0][r][c] = *(const v8h*)kp;
        const v8h vv = *(const v8h*)vp;
        #pragma unroll
        for (int i = 0; i < 8; ++i) Vt[0][c + i][r] = vv[i];
    }
    __syncthreads();

    for (int kt = 0; kt < L_ / 32; ++kt) {
        const int p = kt & 1;
        const int nkt = (kt < L_ / 32 - 1) ? kt + 1 : L_ / 32 - 1;   // clamped
        const size_t noff = (size_t)nkt * 32 * D_;
        const v8h kvn = *(const v8h*)(kp + noff);
        const v8h vvn = *(const v8h*)(vp + noff);

        v8f s0, s1;
        #pragma unroll
        for (int e = 0; e < 8; ++e) { s0[e] = 0.f; s1[e] = 0.f; }
        #pragma unroll
        for (int kc = 0; kc < 2; ++kc) {
            v16h bk0, bk1;
            #pragma unroll
            for (int e = 0; e < 16; ++e) {
                const int kkd = (kc << 5) + (hf << 4) + e;
                bk0[e] = Ks[p][l16][kkd];
                bk1[e] = Ks[p][16 + l16][kkd];
            }
            s0 = __builtin_amdgcn_wmma_f32_16x16x32_f16(false, aq[kc], false, bk0,
                                                        (short)0, s0, false, false);
            s1 = __builtin_amdgcn_wmma_f32_16x16x32_f16(false, aq[kc], false, bk1,
                                                        (short)0, s1, false, false);
        }

        float p0[8], p1[8], alpha[8];
        #pragma unroll
        for (int e = 0; e < 8; ++e) {
            const float a0 = s0[e] * 0.125f;   // 1/sqrt(DK)
            const float a1 = s1[e] * 0.125f;
            float rx = fmaxf(a0, a1);
            #pragma unroll
            for (int m = 1; m <= 8; m <<= 1) rx = fmaxf(rx, __shfl_xor(rx, m, 32));
            const float nm = fmaxf(mrow[e], rx);
            alpha[e] = __expf(mrow[e] - nm);
            mrow[e]  = nm;
            p0[e] = __expf(a0 - nm);
            p1[e] = __expf(a1 - nm);
            float rs = p0[e] + p1[e];
            #pragma unroll
            for (int m = 1; m <= 8; m <<= 1) rs += __shfl_xor(rs, m, 32);
            lrow[e] = lrow[e] * alpha[e] + rs;
        }
        #pragma unroll
        for (int t = 0; t < 4; ++t)
            #pragma unroll
            for (int e = 0; e < 8; ++e) acc[t][e] *= alpha[e];

        // per-wave P relayout; same-wave DS ops are in-order -> no barrier
        #pragma unroll
        for (int e = 0; e < 8; ++e) {
            const int rr = e + (hf << 3);
            Ps[wave][rr][l16]      = (_Float16)p0[e];
            Ps[wave][rr][16 + l16] = (_Float16)p1[e];
        }
        asm volatile("" ::: "memory");   // keep DS program order

        v16h ap;
        #pragma unroll
        for (int e = 0; e < 16; ++e)
            ap[e] = Ps[wave][l16][(hf << 4) + e];
        #pragma unroll
        for (int t = 0; t < 4; ++t) {
            v16h bv;
            #pragma unroll
            for (int e = 0; e < 16; ++e)
                bv[e] = Vt[p][(t << 4) + l16][(hf << 4) + e];
            acc[t] = __builtin_amdgcn_wmma_f32_16x16x32_f16(false, ap, false, bv,
                                                            (short)0, acc[t], false, false);
        }

        *(v8h*)&Ks[p ^ 1][r][c] = kvn;                 // dead on last iter
        #pragma unroll
        for (int i = 0; i < 8; ++i) Vt[p ^ 1][c + i][r] = vvn[i];
        __syncthreads();
    }

    #pragma unroll
    for (int e = 0; e < 8; ++e) {
        const float inv = 1.f / lrow[e];
        const int row = qbase + e + (hf << 3);
        #pragma unroll
        for (int t = 0; t < 4; ++t)
            O[(seqbase + row) * D_ + hd + (t << 4) + l16] = (_Float16)(acc[t][e] * inv);
    }
}

// ---------------------------------------------------------------------------
extern "C" void kernel_launch(void* const* d_in, const int* in_sizes, int n_in,
                              void* d_out, int out_size, void* d_ws, size_t ws_size,
                              hipStream_t stream)
{
    (void)in_sizes; (void)n_in; (void)out_size; (void)ws_size;
    const float* query = (const float*)d_in[0];
    const float* key   = (const float*)d_in[1];
    const float* value = (const float*)d_in[2];
    const float* W0    = (const float*)d_in[3];
    const float* b0    = (const float*)d_in[4];
    const float* Wout  = (const float*)d_in[5];
    const float* bout  = (const float*)d_in[6];
    float* out = (float*)d_out;

    _Float16* ws = (_Float16*)d_ws;
    const size_t N = (size_t)M_ * D_;          // 16.77M halves per buffer
    _Float16* lin_q = ws;                      // reused as attention output
    _Float16* lin_k = ws + N;
    _Float16* lin_v = ws + 2 * N;
    _Float16* qmix  = ws + 3 * N;
    _Float16* kmix  = ws + 4 * N;
    _Float16* wp0   = ws + 5 * N;              // packed W0   (512 KB)
    _Float16* wpo   = wp0 + WPACK_ELEMS;       // packed Wout (512 KB)

    dim3 blk(256);
    k_pack_w<<<64, blk, 0, stream>>>(W0,   wp0);
    k_pack_w<<<64, blk, 0, stream>>>(Wout, wpo);
    k_linear_f32_f16<<<M_ / 32, blk, 0, stream>>>(query, wp0, b0, lin_q);
    k_linear_f32_f16<<<M_ / 32, blk, 0, stream>>>(key,   wp0, b0, lin_k);
    k_linear_f32_f16<<<M_ / 32, blk, 0, stream>>>(value, wp0, b0, lin_v);
    k_local_mix<<<M_ / 8, blk, 0, stream>>>(lin_q, b0, qmix);
    k_local_mix<<<M_ / 8, blk, 0, stream>>>(lin_k, b0, kmix);
    k_flash_attn<<<B_ * H_ * (L_ / 128), blk, 0, stream>>>(qmix, kmix, lin_v, lin_q);
    k_proj_f16_f32<<<M_ / 32, blk, 0, stream>>>(lin_q, wpo, bout, out);
}
